// Attention_50500225466997
// MI455X (gfx1250) — matrix operand order repaired and checked by reading, not compiled, not run on hardware
//
#include <hip/hip_runtime.h>
#include <hip/hip_bf16.h>
#include <math.h>
#include <stdint.h>

// ---------------------------------------------------------------------------
// Types for CDNA5 WMMA (wave32): f16 operands, f32 accumulation.
// ---------------------------------------------------------------------------
typedef _Float16 f16;
typedef __attribute__((ext_vector_type(8)))  _Float16 v8h;
typedef __attribute__((ext_vector_type(16))) _Float16 v16h;
typedef __attribute__((ext_vector_type(8)))  float    v8f;

union V16 { v16h v; v8h h[2]; };

#define B_    16
#define CH_   256
#define NH_   4
#define DH_   64
#define DK_   32
#define L_    1024
#define CQKV_ 512
#define SCALE_ 0.17677669529663687f   // 1/sqrt(DK)

static __device__ __forceinline__ v8f wmma_f16(v16h a, v16h b, v8f c) {
    // D = A(16x32) * B(32x16) + C(16x16), fp32 accumulate
    return __builtin_amdgcn_wmma_f32_16x16x32_f16(false, a, false, b,
                                                  (short)0, c, false, false);
}

// ---------------------------------------------------------------------------
// CDNA5 async global->LDS copy (ASYNCcnt path, no VGPR round-trip).
// VFLAT async form mirrors global_load_b128: vdst = LDS byte offset (VGPR),
// vaddr = 64-bit global address. Generic LDS pointers carry the LDS offset in
// their low 32 bits (SHARED_BASE occupies [63:32]), so truncation is exact.
// ---------------------------------------------------------------------------
static __device__ __forceinline__ void async_ld16(void* lds, const void* g) {
    unsigned           loff = (unsigned)(uintptr_t)lds;
    unsigned long long ga   = (unsigned long long)(uintptr_t)g;
    asm volatile("global_load_async_to_lds_b128 %0, %1, off"
                 :: "v"(loff), "v"(ga) : "memory");
}
static __device__ __forceinline__ void wait_async0() {
    asm volatile("s_wait_asynccnt 0x0" ::: "memory");
}

// ---------------------------------------------------------------------------
// Kernel 0: precision / layout conversion.
//   xbf[b][l][c]  = f16(x[b][c][l])   (transpose so GEMM-B frags are contiguous in c)
//   wqkv_h, wpw_h = f16 copies of the 1x1 conv weights (o-major, contiguous in c)
// ---------------------------------------------------------------------------
__global__ __launch_bounds__(256) void k_convert(const float* __restrict__ x,
                                                 const float* __restrict__ w_qkv,
                                                 const float* __restrict__ w_pw,
                                                 f16* __restrict__ xbf,
                                                 f16* __restrict__ wqkv_h,
                                                 f16* __restrict__ wpw_h) {
    const size_t NX  = (size_t)B_ * CH_ * L_;
    const size_t NW1 = (size_t)CQKV_ * CH_;
    const size_t NW2 = (size_t)CH_ * CH_;
    size_t t = (size_t)blockIdx.x * 256 + threadIdx.x;
    if (t < NX) {
        int    c = (int)(t % CH_);
        size_t r = t / CH_;
        int    l = (int)(r % L_);
        int    b = (int)(r / L_);
        xbf[t] = (f16)x[((size_t)b * CH_ + c) * L_ + l];
    } else if (t < NX + NW1) {
        size_t i = t - NX;
        wqkv_h[i] = (f16)w_qkv[i];
    } else if (t < NX + NW1 + NW2) {
        size_t i = t - NX - NW1;
        wpw_h[i] = (f16)w_pw[i];
    }
}

// ---------------------------------------------------------------------------
// Kernel A: qkv 1x1 conv as WMMA GEMM, BN fused, scatter into attention layouts.
//   QKV[o][l] = sum_c Wqkv[o][c] * X[b][c][l]
//   block = 256 thr (8 waves), tile = 128(o) x 64(l), K-loop 8 x 32.
//   Double-buffered LDS, async global->LDS staging overlapped with WMMA.
// ---------------------------------------------------------------------------
__global__ __launch_bounds__(256) void k_qkv_gemm(const f16* __restrict__ wqkv_h,
                                                  const f16* __restrict__ xbf,
                                                  const float* __restrict__ s_qkv,
                                                  const float* __restrict__ b_qkv,
                                                  f16* __restrict__ qbf,
                                                  f16* __restrict__ kbf,
                                                  f16* __restrict__ vbf) {
    __shared__ __align__(16) f16 Wt[2][128][32];
    __shared__ __align__(16) f16 Xt[2][64][32];
    const int tid  = threadIdx.x;
    const int wave = tid >> 5, lane = tid & 31;
    const int hi   = lane >> 4, ln = lane & 15;
    const int otile = blockIdx.x, ltile = blockIdx.y, b = blockIdx.z;

    const f16* wbase = wqkv_h + (size_t)(otile * 128) * CH_;
    const f16* xbase = xbf + ((size_t)b * L_ + ltile * 64) * CH_;

    float sArr[8], bArr[8];
#pragma unroll
    for (int r = 0; r < 8; ++r) {
        int og = otile * 128 + wave * 16 + r + 8 * hi;
        sArr[r] = s_qkv[og];
        bArr[r] = b_qkv[og];
    }

    const int srow = tid >> 2, scol = (tid & 3) * 8;   // staging coords
    auto stage = [&](int kc, int buf) {
        // A tile: 128 x 32 (two async b128 per thread)
        async_ld16(&Wt[buf][srow][scol],      wbase + (size_t)srow * CH_ + kc * 32 + scol);
        async_ld16(&Wt[buf][srow + 64][scol], wbase + (size_t)(srow + 64) * CH_ + kc * 32 + scol);
        // B tile: 64 x 32 (one async b128 per thread)
        async_ld16(&Xt[buf][srow][scol],      xbase + (size_t)srow * CH_ + kc * 32 + scol);
    };

    stage(0, 0);
    wait_async0();
    __syncthreads();

    v8f acc[4] = {};
    for (int kc = 0; kc < 8; ++kc) {
        const int cur = kc & 1;
        if (kc < 7) stage(kc + 1, cur ^ 1);                       // overlap copy w/ math
        if (kc < 6)                                               // warm L2 two steps out
            __builtin_prefetch(xbase + (size_t)srow * CH_ + (kc + 2) * 32, 0, 1);

        V16 Af;
        const f16* ap = &Wt[cur][wave * 16 + ln][0];
        Af.h[0] = *(const v8h*)(ap + hi * 16);
        Af.h[1] = *(const v8h*)(ap + hi * 16 + 8);
#pragma unroll
        for (int nt = 0; nt < 4; ++nt) {
            V16 Bf;
            const f16* bp = &Xt[cur][nt * 16 + ln][hi * 16];
            Bf.h[0] = *(const v8h*)bp;
            Bf.h[1] = *(const v8h*)(bp + 8);
            acc[nt] = wmma_f16(Af.v, Bf.v, acc[nt]);
        }
        if (kc < 7) { wait_async0(); __syncthreads(); }
    }

    // Epilogue: BN + scatter to q/k/v
#pragma unroll
    for (int nt = 0; nt < 4; ++nt) {
#pragma unroll
        for (int r = 0; r < 8; ++r) {
            int og = otile * 128 + wave * 16 + r + 8 * hi;
            int lg = ltile * 64 + nt * 16 + ln;
            float val = sArr[r] * acc[nt][r] + bArr[r];
            int head = og >> 7, r7 = og & 127;
            if (r7 < DK_)
                qbf[((size_t)(b * NH_ + head) * L_ + lg) * DK_ + r7] = (f16)val;
            else if (r7 < 2 * DK_)
                kbf[((size_t)(b * NH_ + head) * L_ + lg) * DK_ + (r7 - DK_)] = (f16)val;
            else
                vbf[((size_t)(b * NH_ + head) * DH_ + (r7 - 2 * DK_)) * L_ + lg] = (f16)val;
        }
    }
}

// ---------------------------------------------------------------------------
// Kernel B: flash attention (no materialized LxL score matrix).
//   block = 128 thr (4 waves) = 64-query tile; each wave owns 16 query rows.
//   Double-buffered K/V tiles staged with async global->LDS, overlapped with
//   S = Q K^T (4 wmma), online softmax (shfl_xor in 16-lane halves matching
//   the C-layout row ownership), P staged via per-wave LDS (C->A layout),
//   O += P V (8 wmma).
// ---------------------------------------------------------------------------
__global__ __launch_bounds__(128) void k_attn(const f16* __restrict__ qbf,
                                              const f16* __restrict__ kbf,
                                              const f16* __restrict__ vbf,
                                              float* __restrict__ attn_o) {
    __shared__ __align__(16) f16 Kt[2][64][32];    // [key][d]
    __shared__ __align__(16) f16 Vt[2][64][64];    // [d][key]
    __shared__ __align__(16) f16 Pt[4][16][64];    // per-wave [m][key]
    const int tid  = threadIdx.x;
    const int wave = tid >> 5, lane = tid & 31;
    const int hi   = lane >> 4, ln = lane & 15;
    const int qtile = blockIdx.x, bh = blockIdx.y;

    const int kr = tid >> 2, kc8 = (tid & 3) * 8;  // K staging coords
    const int vr = tid >> 3, vc8 = (tid & 7) * 8;  // V staging coords
    auto stage = [&](int jt, int buf) {
        const f16* kb = kbf + ((size_t)bh * L_ + jt * 64) * DK_;
        const f16* vb = vbf + (size_t)bh * DH_ * L_ + jt * 64;
        async_ld16(&Kt[buf][kr][kc8],      kb + (size_t)kr * DK_ + kc8);
        async_ld16(&Kt[buf][kr + 32][kc8], kb + (size_t)(kr + 32) * DK_ + kc8);
#pragma unroll
        for (int i = 0; i < 4; ++i)
            async_ld16(&Vt[buf][vr + 16 * i][vc8], vb + (size_t)(vr + 16 * i) * L_ + vc8);
    };

    // Q A-frag straight from global (DK=32 == one wmma K)
    V16 Qf;
    {
        const f16* qrow = qbf + ((size_t)bh * L_ + qtile * 64 + wave * 16 + ln) * DK_;
        Qf.h[0] = *(const v8h*)(qrow + hi * 16);
        Qf.h[1] = *(const v8h*)(qrow + hi * 16 + 8);
    }

    float m_run[8], l_run[8];
#pragma unroll
    for (int r = 0; r < 8; ++r) { m_run[r] = -1e30f; l_run[r] = 0.f; }
    v8f o_acc[4] = {};
    v8f zf = {};

    stage(0, 0);
    wait_async0();
    __syncthreads();

    for (int jt = 0; jt < 16; ++jt) {
        const int cur = jt & 1;
        if (jt < 15) stage(jt + 1, cur ^ 1);       // async copy behind the math

        // S = Q K^T
        v8f s[4];
#pragma unroll
        for (int nt = 0; nt < 4; ++nt) {
            V16 Bf;
            const f16* bp = &Kt[cur][nt * 16 + ln][hi * 16];
            Bf.h[0] = *(const v8h*)bp;
            Bf.h[1] = *(const v8h*)(bp + 8);
            s[nt] = wmma_f16(Qf.v, Bf.v, zf);
        }

        // online softmax per row (rows r+8*hi live in one 16-lane half)
        float fscale[8];
#pragma unroll
        for (int r = 0; r < 8; ++r) {
            float tmax = -1e30f;
#pragma unroll
            for (int nt = 0; nt < 4; ++nt) {
                s[nt][r] *= SCALE_;
                tmax = fmaxf(tmax, s[nt][r]);
            }
#pragma unroll
            for (int off = 1; off < 16; off <<= 1)
                tmax = fmaxf(tmax, __shfl_xor(tmax, off));
            float m_new = fmaxf(m_run[r], tmax);
            fscale[r] = __expf(m_run[r] - m_new);
            float lsum = 0.f;
#pragma unroll
            for (int nt = 0; nt < 4; ++nt) {
                float p = __expf(s[nt][r] - m_new);
                s[nt][r] = p;
                lsum += p;
            }
#pragma unroll
            for (int off = 1; off < 16; off <<= 1)
                lsum += __shfl_xor(lsum, off);
            l_run[r] = l_run[r] * fscale[r] + lsum;
            m_run[r] = m_new;
        }

        // stage P (C layout) -> LDS rows (A layout source)
#pragma unroll
        for (int nt = 0; nt < 4; ++nt)
#pragma unroll
            for (int r = 0; r < 8; ++r)
                Pt[wave][r + 8 * hi][nt * 16 + ln] = (f16)s[nt][r];
        __syncthreads();

        // rescale running O, then O += P * V
#pragma unroll
        for (int nt = 0; nt < 4; ++nt)
#pragma unroll
            for (int r = 0; r < 8; ++r)
                o_acc[nt][r] *= fscale[r];
#pragma unroll
        for (int kk = 0; kk < 2; ++kk) {
            V16 Pf;
            const f16* pp = &Pt[wave][ln][kk * 32 + hi * 16];
            Pf.h[0] = *(const v8h*)pp;
            Pf.h[1] = *(const v8h*)(pp + 8);
#pragma unroll
            for (int nt = 0; nt < 4; ++nt) {
                V16 Vf;
                const f16* vp = &Vt[cur][nt * 16 + ln][kk * 32 + hi * 16];
                Vf.h[0] = *(const v8h*)vp;
                Vf.h[1] = *(const v8h*)(vp + 8);
                o_acc[nt] = wmma_f16(Pf.v, Vf.v, o_acc[nt]);
            }
        }
        if (jt < 15) { wait_async0(); __syncthreads(); }
    }

    // normalize + store (b, c=h*DH+d, l) fp32
    const int bb = bh >> 2, hh = bh & 3;
#pragma unroll
    for (int nt = 0; nt < 4; ++nt) {
#pragma unroll
        for (int r = 0; r < 8; ++r) {
            float inv = 1.f / l_run[r];
            int d = nt * 16 + ln;
            int qr = qtile * 64 + wave * 16 + r + 8 * hi;
            attn_o[((size_t)bb * CH_ + hh * DH_ + d) * L_ + qr] = o_acc[nt][r] * inv;
        }
    }
}

// ---------------------------------------------------------------------------
// Kernel C: 3x3 depthwise conv on V (zero-pad SAME) + BN + add attention out,
// written as u[b][l][c] f16 for the pointwise GEMM.
// vbf flat index (b*NH+h)*DH+d == b*CH + c, so vbf doubles as (b,c,l).
// ---------------------------------------------------------------------------
__global__ __launch_bounds__(256) void k_dwconv_add(const f16* __restrict__ vbf,
                                                    const float* __restrict__ attn_o,
                                                    const float* __restrict__ w_dw,
                                                    const float* __restrict__ s_dw,
                                                    const float* __restrict__ b_dw,
                                                    f16* __restrict__ u) {
    __shared__ float plane[32][32];
    const int bc = blockIdx.x;            // b*CH + c
    const int b = bc / CH_, c = bc % CH_;
    const int tid = threadIdx.x;
    for (int i = tid; i < 1024; i += 256)
        plane[i >> 5][i & 31] = (float)vbf[(size_t)bc * L_ + i];
    __syncthreads();
    float w9[9];
#pragma unroll
    for (int k = 0; k < 9; ++k) w9[k] = w_dw[c * 9 + k];
    const float sd = s_dw[c], bd = b_dw[c];
    for (int i = tid; i < 1024; i += 256) {
        int y = i >> 5, xc = i & 31;
        float acc = 0.f;
#pragma unroll
        for (int ky = -1; ky <= 1; ++ky)
#pragma unroll
            for (int kx = -1; kx <= 1; ++kx) {
                int yy = y + ky, xx = xc + kx;
                if (yy >= 0 && yy < 32 && xx >= 0 && xx < 32)
                    acc += w9[(ky + 1) * 3 + kx + 1] * plane[yy][xx];
            }
        float val = attn_o[(size_t)bc * L_ + i] + sd * acc + bd;
        u[((size_t)b * L_ + i) * CH_ + c] = (f16)val;
    }
}

// ---------------------------------------------------------------------------
// Kernel D: pointwise 1x1 conv as WMMA GEMM + BN, fp32 output (b, o, l).
// Same double-buffered async-staged structure as kernel A.
// ---------------------------------------------------------------------------
__global__ __launch_bounds__(256) void k_pw_gemm(const f16* __restrict__ wpw_h,
                                                 const f16* __restrict__ u,
                                                 const float* __restrict__ s_pw,
                                                 const float* __restrict__ b_pw,
                                                 float* __restrict__ out) {
    __shared__ __align__(16) f16 Wt[2][128][32];
    __shared__ __align__(16) f16 Xt[2][64][32];
    const int tid  = threadIdx.x;
    const int wave = tid >> 5, lane = tid & 31;
    const int hi   = lane >> 4, ln = lane & 15;
    const int otile = blockIdx.x, ltile = blockIdx.y, b = blockIdx.z;

    const f16* wbase = wpw_h + (size_t)(otile * 128) * CH_;
    const f16* xbase = u + ((size_t)b * L_ + ltile * 64) * CH_;

    float sArr[8], bArr[8];
#pragma unroll
    for (int r = 0; r < 8; ++r) {
        int og = otile * 128 + wave * 16 + r + 8 * hi;
        sArr[r] = s_pw[og];
        bArr[r] = b_pw[og];
    }

    const int srow = tid >> 2, scol = (tid & 3) * 8;
    auto stage = [&](int kc, int buf) {
        async_ld16(&Wt[buf][srow][scol],      wbase + (size_t)srow * CH_ + kc * 32 + scol);
        async_ld16(&Wt[buf][srow + 64][scol], wbase + (size_t)(srow + 64) * CH_ + kc * 32 + scol);
        async_ld16(&Xt[buf][srow][scol],      xbase + (size_t)srow * CH_ + kc * 32 + scol);
    };

    stage(0, 0);
    wait_async0();
    __syncthreads();

    v8f acc[4] = {};
    for (int kc = 0; kc < 8; ++kc) {
        const int cur = kc & 1;
        if (kc < 7) stage(kc + 1, cur ^ 1);
        if (kc < 6)
            __builtin_prefetch(xbase + (size_t)srow * CH_ + (kc + 2) * 32, 0, 1);

        V16 Af;
        const f16* ap = &Wt[cur][wave * 16 + ln][0];
        Af.h[0] = *(const v8h*)(ap + hi * 16);
        Af.h[1] = *(const v8h*)(ap + hi * 16 + 8);
#pragma unroll
        for (int nt = 0; nt < 4; ++nt) {
            V16 Bf;
            const f16* bp = &Xt[cur][nt * 16 + ln][hi * 16];
            Bf.h[0] = *(const v8h*)bp;
            Bf.h[1] = *(const v8h*)(bp + 8);
            acc[nt] = wmma_f16(Af.v, Bf.v, acc[nt]);
        }
        if (kc < 7) { wait_async0(); __syncthreads(); }
    }

#pragma unroll
    for (int nt = 0; nt < 4; ++nt) {
#pragma unroll
        for (int r = 0; r < 8; ++r) {
            int og = otile * 128 + wave * 16 + r + 8 * hi;
            int lg = ltile * 64 + nt * 16 + ln;
            out[((size_t)b * CH_ + og) * L_ + lg] = sArr[r] * acc[nt][r] + bArr[r];
        }
    }
}

// ---------------------------------------------------------------------------
// Launch
// ---------------------------------------------------------------------------
extern "C" void kernel_launch(void* const* d_in, const int* in_sizes, int n_in,
                              void* d_out, int out_size, void* d_ws, size_t ws_size,
                              hipStream_t stream) {
    (void)in_sizes; (void)n_in; (void)out_size; (void)ws_size;
    const float* x     = (const float*)d_in[0];
    const float* w_qkv = (const float*)d_in[1];
    const float* s_qkv = (const float*)d_in[2];
    const float* b_qkv = (const float*)d_in[3];
    const float* w_dw  = (const float*)d_in[4];
    const float* s_dw  = (const float*)d_in[5];
    const float* b_dw  = (const float*)d_in[6];
    const float* w_pw  = (const float*)d_in[7];
    const float* s_pw  = (const float*)d_in[8];
    const float* b_pw  = (const float*)d_in[9];
    float* out = (float*)d_out;

    // workspace carve-up (~50 MB; everything stays L2-resident on MI455X)
    char* ws = (char*)d_ws;
    size_t off = 0;
    auto carve = [&](size_t bytes) -> void* {
        void* p = ws + off;
        off = (off + bytes + 255) & ~(size_t)255;
        return p;
    };
    f16*   xbf    = (f16*)carve((size_t)B_ * L_ * CH_ * 2);
    f16*   wqkv_h = (f16*)carve((size_t)CQKV_ * CH_ * 2);
    f16*   wpw_h  = (f16*)carve((size_t)CH_ * CH_ * 2);
    f16*   qbf    = (f16*)carve((size_t)B_ * NH_ * L_ * DK_ * 2);
    f16*   kbf    = (f16*)carve((size_t)B_ * NH_ * L_ * DK_ * 2);
    f16*   vbf    = (f16*)carve((size_t)B_ * NH_ * DH_ * L_ * 2);
    float* attn_o = (float*)carve((size_t)B_ * CH_ * L_ * 4);
    f16*   u      = (f16*)carve((size_t)B_ * L_ * CH_ * 2);

    {
        size_t total = (size_t)B_ * CH_ * L_ + (size_t)CQKV_ * CH_ + (size_t)CH_ * CH_;
        int blocks = (int)((total + 255) / 256);
        k_convert<<<blocks, 256, 0, stream>>>(x, w_qkv, w_pw, xbf, wqkv_h, wpw_h);
    }
    k_qkv_gemm<<<dim3(4, 16, 16), 256, 0, stream>>>(wqkv_h, xbf, s_qkv, b_qkv, qbf, kbf, vbf);
    k_attn<<<dim3(16, 64), 128, 0, stream>>>(qbf, kbf, vbf, attn_o);
    k_dwconv_add<<<B_ * CH_, 256, 0, stream>>>(vbf, attn_o, w_dw, s_dw, b_dw, u);
    k_pw_gemm<<<dim3(2, 16, 16), 256, 0, stream>>>(wpw_h, u, s_pw, b_pw, out);
}
